// SlidingWindowSelfAttention_16097537425813
// MI455X (gfx1250) — matrix operand order repaired and checked by reading, not compiled, not run on hardware
//
#include <hip/hip_runtime.h>

// ---------------- problem constants ----------------
constexpr int B_  = 2;
constexpr int S_  = 2048;
constexpr int D_  = 1024;
constexpr int H_  = 16;
constexpr int HD_ = 64;
constexpr int R_  = 8;
constexpr int HALF_ = 128;
constexpr int T_  = S_ + R_;   // 2056
constexpr int TP_ = 2080;      // T padded to multiple of 32 (key-chunk size)

typedef __bf16 bf16;
typedef __attribute__((ext_vector_type(8)))  bf16  v8bf;
typedef __attribute__((ext_vector_type(16))) bf16  v16bf;
typedef __attribute__((ext_vector_type(8)))  float v8f;

// ---------------- WMMA fragment helpers ----------------
// A-fragment (16x32 bf16, MxK): lane holds row m0+(lane&15); two contiguous
// 8-element chunks at K = k0 + {0|8} and +16 (per CDNA5 16-bit A layout).
__device__ inline v16bf load_a_frag(const bf16* __restrict__ base, int row_stride,
                                    int m0, int k0, int lane) {
  int row = m0 + (lane & 15);
  int kk  = k0 + ((lane >> 4) << 3);
  const bf16* p = base + (size_t)row * row_stride + kk;
  v8bf c0 = *(const v8bf*)(p);
  v8bf c1 = *(const v8bf*)(p + 16);
  return __builtin_shufflevector(c0, c1, 0,1,2,3,4,5,6,7,8,9,10,11,12,13,14,15);
}

// B-fragment (32x16 bf16, KxN) from a K-contiguous source: lane holds column
// n0+(lane&15); two contiguous 8-element chunks at K = k0 + {0|8} and +16.
__device__ inline v16bf load_b_frag(const bf16* __restrict__ base, int row_stride,
                                    int n0, int k0, int lane) {
  int col = n0 + (lane & 15);
  int kk  = k0 + ((lane >> 4) << 3);
  const bf16* p = base + (size_t)col * row_stride + kk;
  v8bf c0 = *(const v8bf*)(p);
  v8bf c1 = *(const v8bf*)(p + 16);
  return __builtin_shufflevector(c0, c1, 0,1,2,3,4,5,6,7,8,9,10,11,12,13,14,15);
}

__device__ inline v8f wmma_bf16(v16bf a, v16bf b, v8f c) {
  return __builtin_amdgcn_wmma_f32_16x16x32_bf16(false, a, false, b, (short)0, c,
                                                 false, false);
}

__device__ inline float rowmax16(float v) {
  #pragma unroll
  for (int m = 1; m < 16; m <<= 1) v = fmaxf(v, __shfl_xor(v, m, 16));
  return v;
}
__device__ inline float rowsum16(float v) {
  #pragma unroll
  for (int m = 1; m < 16; m <<= 1) v += __shfl_xor(v, m, 16);
  return v;
}

// ---------------- prep kernels ----------------
__global__ void build_xr_kernel(const float* __restrict__ x,
                                const float* __restrict__ ada,
                                const float* __restrict__ regs,
                                bf16* __restrict__ xr) {
  int idx = blockIdx.x * blockDim.x + threadIdx.x;
  if (idx >= B_ * TP_ * D_) return;
  int d = idx % D_;
  int t = (idx / D_) % TP_;
  int b = idx / (D_ * TP_);
  float v;
  if (t < R_)      v = regs[t * D_ + d] * (1.0f + ada[b * D_ + d]);
  else if (t < T_) v = x[((size_t)b * S_ + (t - R_)) * D_ + d];
  else             v = 0.0f;
  xr[idx] = (bf16)v;
}

__global__ void cast_f32_bf16_kernel(const float* __restrict__ in,
                                     bf16* __restrict__ out, int n) {
  int idx = blockIdx.x * blockDim.x + threadIdx.x;
  if (idx < n) out[idx] = (bf16)in[idx];
}

// ---------------- QKV projection: qkv = xr @ Wqkv^T + b ----------------
// M = B*TP_ rows, N = 3*D, K = D. Wave computes 16x64; scatters into
// Q [b,h,t,hd] (pre-scaled by 1/8), K [b,h,t,hd], V transposed [b,h,hd,t].
__global__ __launch_bounds__(256) void qkv_gemm_kernel(
    const bf16* __restrict__ xr, const bf16* __restrict__ w,
    const float* __restrict__ bias, bf16* __restrict__ Qo,
    bf16* __restrict__ Ko, bf16* __restrict__ Vt) {
  const int lane = threadIdx.x & 31;
  const int wv   = threadIdx.x >> 5;
  const int strips = (B_ * TP_) / 16;            // 260
  int strip = blockIdx.x * 8 + wv;
  if (strip >= strips) return;                   // wave-uniform, no barriers
  const int m0 = strip * 16;
  const int n0 = blockIdx.y * 64;

  v8f acc[4];
  #pragma unroll
  for (int j = 0; j < 4; ++j) acc[j] = (v8f){0,0,0,0,0,0,0,0};

  for (int k = 0; k < D_; k += 32) {
    v16bf a = load_a_frag(xr, D_, m0, k, lane);
    #pragma unroll
    for (int jj = 0; jj < 4; ++jj) {
      v16bf bw = load_b_frag(w, D_, n0 + jj * 16, k, lane);
      acc[jj] = wmma_bf16(a, bw, acc[jj]);
    }
  }

  const int which = n0 >> 10;                    // 0=Q, 1=K, 2=V (uniform/block)
  const int half8 = (lane >> 4) << 3;
  #pragma unroll
  for (int jj = 0; jj < 4; ++jj) {
    int n    = n0 + jj * 16 + (lane & 15);
    float bv = bias[n];
    int dcol = n & (D_ - 1);
    int h  = dcol >> 6;
    int hd = dcol & (HD_ - 1);
    #pragma unroll
    for (int i = 0; i < 8; ++i) {
      int m = m0 + i + half8;
      int b = m / TP_, t = m % TP_;
      float v = acc[jj][i] + bv;
      if (which == 0)
        Qo[((size_t)(b * H_ + h) * TP_ + t) * HD_ + hd] = (bf16)(v * 0.125f);
      else if (which == 1)
        Ko[((size_t)(b * H_ + h) * TP_ + t) * HD_ + hd] = (bf16)v;
      else
        Vt[((size_t)(b * H_ + h) * HD_ + hd) * TP_ + t] = (bf16)v;
    }
  }
}

// ---------------- flash attention, one 16-query tile per wave ----------------
__global__ __launch_bounds__(256) void attn_kernel(
    const bf16* __restrict__ Q, const bf16* __restrict__ K,
    const bf16* __restrict__ Vt, bf16* __restrict__ attn) {
  __shared__ __align__(64) bf16 pbuf[8][16][32];   // P transpose staging, 1KB/wave
  const int lane = threadIdx.x & 31;
  const int wv   = threadIdx.x >> 5;
  const int QT   = S_ / 16;                        // 128 query tiles per (b,h)
  int tile = blockIdx.x * 8 + wv;
  int b  = tile / (H_ * QT);
  int rm = tile % (H_ * QT);
  int h  = rm / QT;
  int p0 = (rm % QT) * 16;

  const bf16* Qh = Q  + (size_t)(b * H_ + h) * TP_ * HD_;
  const bf16* Kh = K  + (size_t)(b * H_ + h) * TP_ * HD_;
  const bf16* Vh = Vt + (size_t)(b * H_ + h) * HD_ * TP_;

  // Q fragments for rows t = R + p0 + (lane&15), hd split 0..31 / 32..63
  v16bf aq0 = load_a_frag(Qh, HD_, R_ + p0, 0,  lane);
  v16bf aq1 = load_a_frag(Qh, HD_, R_ + p0, 32, lane);

  float mrow[8], lrow[8];
  v8f o[4];
  #pragma unroll
  for (int i = 0; i < 8; ++i) { mrow[i] = -3.0e38f; lrow[i] = 0.0f; }
  #pragma unroll
  for (int j = 0; j < 4; ++j) o[j] = (v8f){0,0,0,0,0,0,0,0};

  const int half8 = (lane >> 4) << 3;
  int lo_t = p0 - (HALF_ - R_);  if (lo_t < 0)      lo_t = 0;
  int hi_t = p0 + 15 + HALF_ + R_; if (hi_t > T_-1) hi_t = T_ - 1;
  const int c_lo = lo_t >> 5, c_hi = hi_t >> 5;

  for (int phase = 0; phase < 2; ++phase) {
    // phase 0: register chunk (t=0..31) if not already covered; phase 1: window
    int cb = (phase == 0) ? 0 : c_lo;
    int ce = (phase == 0) ? ((c_lo > 0) ? 0 : -1) : c_hi;
    for (int c = cb; c <= ce; ++c) {
      const int t0 = c << 5;
      __builtin_prefetch(Kh + (size_t)(t0 + 32) * HD_, 0, 0);  // next key chunk

      // scores: two 16x16 tiles over 32 keys, K-dim = HD (2 wmma each)
      v16bf bk0a = load_b_frag(Kh, HD_, t0,      0,  lane);
      v16bf bk0b = load_b_frag(Kh, HD_, t0,      32, lane);
      v16bf bk1a = load_b_frag(Kh, HD_, t0 + 16, 0,  lane);
      v16bf bk1b = load_b_frag(Kh, HD_, t0 + 16, 32, lane);
      v8f z = (v8f){0,0,0,0,0,0,0,0};
      v8f s0 = wmma_bf16(aq0, bk0a, z); s0 = wmma_bf16(aq1, bk0b, s0);
      v8f s1 = wmma_bf16(aq0, bk1a, z); s1 = wmma_bf16(aq1, bk1b, s1);

      const int tA = t0 + (lane & 15), tB = tA + 16;
      #pragma unroll
      for (int i = 0; i < 8; ++i) {
        int p = p0 + i + half8;
        float x0 = s0[i], x1 = s1[i];
        // mask: registers always visible; tokens inside +/-HALF window
        int pkA = tA - R_, pkB = tB - R_;
        bool vA = (tA < R_) || (pkA < S_ && abs(p - pkA) <= HALF_);
        bool vB = (tB < R_) || (pkB < S_ && abs(p - pkB) <= HALF_);
        if (!vA) x0 = -1.0e30f;
        if (!vB) x1 = -1.0e30f;
        float mx   = rowmax16(fmaxf(x0, x1));
        float mnew = fmaxf(mrow[i], mx);
        float sf   = __expf(mrow[i] - mnew);
        float e0 = __expf(x0 - mnew), e1 = __expf(x1 - mnew);
        lrow[i] = lrow[i] * sf + rowsum16(e0 + e1);
        mrow[i] = mnew;
        o[0][i] *= sf; o[1][i] *= sf; o[2][i] *= sf; o[3][i] *= sf;
        int r = i + half8;
        pbuf[wv][r][lane & 15]        = (bf16)e0;   // C-layout -> row-major LDS
        pbuf[wv][r][16 + (lane & 15)] = (bf16)e1;
      }
      // same-wave DS ops are in-order; just wait for store data visibility
      asm volatile("s_wait_dscnt 0" ::: "memory");
      v16bf ap = load_a_frag(&pbuf[wv][0][0], 32, 0, 0, lane);  // P as A-frag

      #pragma unroll
      for (int jj = 0; jj < 4; ++jj) {                // P(16x32keys) x V(32x16hd)
        v16bf bv = load_b_frag(Vh, TP_, jj * 16, t0, lane);
        o[jj] = wmma_bf16(ap, bv, o[jj]);
      }
    }
  }

  #pragma unroll
  for (int jj = 0; jj < 4; ++jj) {
    #pragma unroll
    for (int i = 0; i < 8; ++i) {
      int p = p0 + i + half8;
      float v = o[jj][i] / lrow[i];
      attn[((size_t)(b * S_ + p)) * D_ + h * HD_ + jj * 16 + (lane & 15)] = (bf16)v;
    }
  }
}

// ---------------- output projection: out = attn @ Wout^T + b_out ----------------
__global__ __launch_bounds__(256) void out_gemm_kernel(
    const bf16* __restrict__ ain, const bf16* __restrict__ w,
    const float* __restrict__ bias, float* __restrict__ out) {
  const int lane = threadIdx.x & 31;
  const int wv   = threadIdx.x >> 5;
  const int m0 = (blockIdx.x * 8 + wv) * 16;     // M = 4096, exact fit
  const int n0 = blockIdx.y * 64;

  v8f acc[4];
  #pragma unroll
  for (int j = 0; j < 4; ++j) acc[j] = (v8f){0,0,0,0,0,0,0,0};

  for (int k = 0; k < D_; k += 32) {
    v16bf a = load_a_frag(ain, D_, m0, k, lane);
    #pragma unroll
    for (int jj = 0; jj < 4; ++jj) {
      v16bf bw = load_b_frag(w, D_, n0 + jj * 16, k, lane);
      acc[jj] = wmma_bf16(a, bw, acc[jj]);
    }
  }

  const int half8 = (lane >> 4) << 3;
  #pragma unroll
  for (int jj = 0; jj < 4; ++jj) {
    int n = n0 + jj * 16 + (lane & 15);
    float bv = bias[n];
    #pragma unroll
    for (int i = 0; i < 8; ++i) {
      int m = m0 + i + half8;
      out[(size_t)m * D_ + n] = acc[jj][i] + bv;
    }
  }
}

// ---------------- host launcher ----------------
extern "C" void kernel_launch(void* const* d_in, const int* /*in_sizes*/, int /*n_in*/,
                              void* d_out, int /*out_size*/, void* d_ws,
                              size_t /*ws_size*/, hipStream_t stream) {
  const float* x    = (const float*)d_in[0];
  const float* ada  = (const float*)d_in[1];
  const float* regs = (const float*)d_in[2];
  const float* Wqkv = (const float*)d_in[3];
  const float* bqkv = (const float*)d_in[4];
  const float* Wout = (const float*)d_in[5];
  const float* bout = (const float*)d_in[6];
  float* out = (float*)d_out;

  char* ws = (char*)d_ws;
  size_t off = 0;
  bf16* xr  = (bf16*)(ws + off); off += (size_t)B_ * TP_ * D_ * 2;        // 8.52 MB
  bf16* wq  = (bf16*)(ws + off); off += (size_t)3 * D_ * D_ * 2;          // 6.29 MB
  bf16* wo  = (bf16*)(ws + off); off += (size_t)D_ * D_ * 2;              // 2.10 MB
  bf16* Qb  = (bf16*)(ws + off); off += (size_t)B_ * H_ * TP_ * HD_ * 2;  // 8.52 MB
  bf16* Kb  = (bf16*)(ws + off); off += (size_t)B_ * H_ * TP_ * HD_ * 2;  // 8.52 MB
  bf16* Vtb = (bf16*)(ws + off); off += (size_t)B_ * H_ * HD_ * TP_ * 2;  // 8.52 MB
  bf16* att = (bf16*)(ws + off); off += (size_t)B_ * S_ * D_ * 2;         // 8.39 MB
  // total ~50.9 MB of workspace, all L2-resident on MI455X (192 MB L2)

  int nxr = B_ * TP_ * D_;
  build_xr_kernel<<<(nxr + 255) / 256, 256, 0, stream>>>(x, ada, regs, xr);
  cast_f32_bf16_kernel<<<(3 * D_ * D_ + 255) / 256, 256, 0, stream>>>(Wqkv, wq, 3 * D_ * D_);
  cast_f32_bf16_kernel<<<(D_ * D_ + 255) / 256, 256, 0, stream>>>(Wout, wo, D_ * D_);

  dim3 g1(((B_ * TP_) / 16 + 7) / 8, (3 * D_) / 64);   // (33, 48)
  qkv_gemm_kernel<<<g1, 256, 0, stream>>>(xr, wq, bqkv, Qb, Kb, Vtb);

  int tiles = B_ * H_ * (S_ / 16);                     // 4096 query tiles
  attn_kernel<<<tiles / 8, 256, 0, stream>>>(Qb, Kb, Vtb, att);

  dim3 g2((B_ * S_) / 128, D_ / 64);                   // (32, 16)
  out_gemm_kernel<<<g2, 256, 0, stream>>>(att, wo, bout, out);
}
